// DNAMite_19756849561696
// MI455X (gfx1250) — matrix operand order repaired and checked by reading, not compiled, not run on hardware
//
#include <hip/hip_runtime.h>

// ---------------------------------------------------------------------------
// DNAMite grouped-MLP forward on gfx1250 (CDNA5).
//   main path:  64 groups,  MLP 32 -> 128 -> 128 -> 1
//   pair path: 2016 groups, MLP 64 -> 128 -> 128 -> 1
// Strategy:
//   * prepass: convert/transpose all weights fp32 -> bf16 into d_ws, laid out
//     as the exact padded LDS image per group; convert embeddings to bf16.
//   * GEMM kernel: one block per (group, 128-row batch tile); weights staged
//     into LDS with GLOBAL_LOAD_ASYNC_TO_LDS_B128; all matmuls via
//     v_wmma_f32_16x16x32_bf16; layer-2 dot + smooth_z gate + atomicAdd.
//   * grid = (batch-tiles, groups) so the 4 blocks sharing one group's weight
//     image are adjacent in dispatch order -> weight re-reads hit L2.
// ---------------------------------------------------------------------------

#define BATCH   512
#define BM      128          // batch rows per block
#define NF      64
#define BINS    64
#define NE      32
#define NH      128
#define NPAIR   2016
#define EMB_ROWS 4096

typedef __bf16 bf16;
typedef __attribute__((ext_vector_type(16))) __bf16 v16bf;
typedef __attribute__((ext_vector_type(8)))  __bf16 v8bf;
typedef __attribute__((ext_vector_type(8)))  float  v8f;

// ---------------- async global -> LDS helpers (CDNA5) ----------------------
#if __has_builtin(__builtin_amdgcn_global_load_async_to_lds_b128)
#define HAS_ASYNC_LDS 1
typedef int v4i_t __attribute__((vector_size(16)));
typedef __attribute__((address_space(1))) v4i_t gv4i_t;   // global
typedef __attribute__((address_space(3))) v4i_t lv4i_t;   // LDS
__device__ __forceinline__ void async_copy16(const void* g, void* l) {
    __builtin_amdgcn_global_load_async_to_lds_b128((gv4i_t*)g, (lv4i_t*)l, 0, 0);
}
#else
#define HAS_ASYNC_LDS 0
__device__ __forceinline__ void async_copy16(const void* g, void* l) {
    *(v8bf*)l = *(const v8bf*)g;   // fallback: load+store through VGPRs
}
#endif

__device__ __forceinline__ void wait_async_copies() {
#if HAS_ASYNC_LDS
#if __has_builtin(__builtin_amdgcn_s_wait_asynccnt)
    __builtin_amdgcn_s_wait_asynccnt(0);
#else
    asm volatile("s_wait_asynccnt 0x0" ::: "memory");
#endif
#endif
}

// ---------------------------------------------------------------------------
__device__ __forceinline__ float smooth_z(float z) {
    const float g = 1.0f;
    float s = (-2.0f / (g * g * g)) * z * z * z + (3.0f / (2.0f * g)) * z + 0.5f;
    return (z <= -0.5f * g) ? 0.0f : ((z >= 0.5f * g) ? 1.0f : s);
}

// One GEMM layer: [BM x K] (bf16, LDS) x [K x NH] (bf16, LDS, transposed as
// Wt[n][k], K-stride LDK) + bias, ReLU -> [BM x NH] bf16 in LDS.
// Wave `wave` owns rows [wave*16, wave*16+16).
template <int K, int LDK>
__device__ __forceinline__ void gemm_layer(const bf16* __restrict__ sIn, int ldin,
                                           const bf16* __restrict__ sWt,
                                           const float* __restrict__ sBias,
                                           bf16* __restrict__ sOut, int ldout,
                                           int wave, int lane) {
    constexpr int KS = K / 32;
    const int mbase = wave * 16;
    const int h    = lane >> 4;    // lane half (0/1)
    const int nlo  = lane & 15;

    // A fragments (16x32 bf16): lane l<16 -> row l, K halves [0..7]+[16..23];
    // lanes 16..31 -> same rows, K halves shifted by 8.
    v16bf aF[KS];
#pragma unroll
    for (int ks = 0; ks < KS; ++ks) {
        const bf16* pa = sIn + (mbase + nlo) * ldin + ks * 32 + h * 8;
        union { v16bf v; v8bf p[2]; } u;
        u.p[0] = *(const v8bf*)(pa);
        u.p[1] = *(const v8bf*)(pa + 16);
        aF[ks] = u.v;
    }

#pragma unroll
    for (int nt = 0; nt < NH / 16; ++nt) {
        const int nb = nt * 16;
        const float bv = sBias[nb + nlo];
        v8f acc = {bv, bv, bv, bv, bv, bv, bv, bv};
#pragma unroll
        for (int ks = 0; ks < KS; ++ks) {
            // B fragment (32x16): lanes 0-15 cover K [0..7]+[16..23], 16-31 cover [8..15]+[24..31].
            const bf16* pb = sWt + (nb + nlo) * LDK + ks * 32 + h * 8;
            union { v16bf v; v8bf p[2]; } u;
            u.p[0] = *(const v8bf*)(pb);
            u.p[1] = *(const v8bf*)(pb + 16);
            acc = __builtin_amdgcn_wmma_f32_16x16x32_bf16(
                false, aF[ks], false, u.v, (short)0, acc, false, false);
        }
        // C/D layout: vgpr r, lane -> row = r + 8*h, col = nlo
#pragma unroll
        for (int r = 0; r < 8; ++r) {
            float v = fmaxf(acc[r], 0.0f);   // ReLU
            sOut[(mbase + r + 8 * h) * ldout + nb + nlo] = (bf16)v;
        }
    }
}

// ---------------------------------------------------------------------------
// Main grouped kernel. WS=true: weights/embeddings pre-converted bf16 in ws,
// staged into LDS by async copies. WS=false: convert fp32 in-kernel.
// grid.x = batch tile, grid.y = group.
template <int K0, bool PAIR, bool WS>
__global__ __launch_bounds__(256) void nam_group_kernel(
    const int*   __restrict__ mains,       // [B, F]
    const int*   __restrict__ pairs,       // [B, P, 2]
    const int*   __restrict__ pairs_list,  // [P, 2]
    const float* __restrict__ emb,         // [4096, 32] fp32 (WS=false)
    const bf16*  __restrict__ embb,        // [4096, 32] bf16 (WS=true)
    const float* __restrict__ w0,          // [G, K0, NH] fp32 (WS=false)
    const float* __restrict__ w1,          // [G, NH, NH] fp32 (WS=false)
    const bf16*  __restrict__ wt0,         // [G][NH][K0+8] bf16 (WS=true)
    const bf16*  __restrict__ wt1,         // [G][NH][NH+8] bf16 (WS=true)
    const float* __restrict__ w2,          // [G, NH, 1]
    const float* __restrict__ b0,          // [G, NH]
    const float* __restrict__ b1,          // [G, NH]
    const float* __restrict__ b2,          // [G, 1]
    const float* __restrict__ z,           // [G]
    float*       __restrict__ out)         // [B]
{
    constexpr int LDK0 = K0 + 8;   // padded K stride for transposed W0
    constexpr int LDK1 = NH + 8;   // padded K stride for transposed W1
    constexpr int SEG  = K0 / 32;  // 32-wide embedding segments per row

    const int g       = blockIdx.y;
    const int rowBase = blockIdx.x * BM;
    const int tid     = threadIdx.x;
    const int wave    = tid >> 5;
    const int lane    = tid & 31;

    __shared__ bf16  sA[BM][K0];         // layer-0 input (gathered embeddings)
    __shared__ bf16  sH0[BM][NH];
    __shared__ bf16  sH1[BM][NH];
    __shared__ bf16  sW0[NH][LDK0];      // w0 transposed: [n][k]
    __shared__ bf16  sW1[NH][LDK1];      // w1 transposed: [n][k]
    __shared__ float sB0[NH];
    __shared__ float sB1[NH];
    __shared__ float sW2[NH];
    __shared__ float sMisc[2];           // {b2, smooth_z(z)}
    __shared__ int   sIdx[BM][2];

    if (tid == 0) {
        sMisc[0] = b2[g];
        sMisc[1] = smooth_z(z[g]);
    }
    for (int i = tid; i < NH; i += 256) {
        sB0[i] = b0[g * NH + i];
        sB1[i] = b1[g * NH + i];
        sW2[i] = w2[g * NH + i];
    }

    // Embedding row indices for this block's batch rows.
    if (PAIR) {
        for (int i = tid; i < BM * 2; i += 256) {
            int r = i >> 1, s = i & 1;
            int b = rowBase + r;
            int f = pairs_list[g * 2 + s];
            sIdx[r][s] = pairs[((size_t)b * NPAIR + g) * 2 + s] + f * BINS;
        }
    } else {
        for (int i = tid; i < BM; i += 256) {
            int b = rowBase + i;
            sIdx[i][0] = mains[b * NF + g] + g * BINS;
        }
    }

    if (WS) {
        // Weights: byte-exact async copy of the pre-transposed bf16 LDS image.
        constexpr int W0B = NH * LDK0 * 2;
        constexpr int W1B = NH * LDK1 * 2;
        const char* g0 = (const char*)(wt0 + (size_t)g * NH * LDK0);
        const char* g1 = (const char*)(wt1 + (size_t)g * NH * LDK1);
        char* l0 = (char*)&sW0[0][0];
        char* l1 = (char*)&sW1[0][0];
        for (int off = tid * 16; off < W0B; off += 256 * 16)
            async_copy16(g0 + off, l0 + off);
        for (int off = tid * 16; off < W1B; off += 256 * 16)
            async_copy16(g1 + off, l1 + off);
        __syncthreads();                 // sIdx now visible to all waves
        // Embedding gather: one 64B row-segment per (row, slot), async B128 x4.
        for (int i = tid; i < BM * SEG; i += 256) {
            int r = i / SEG, s = i % SEG;
            const char* src = (const char*)(embb + (size_t)sIdx[r][s] * NE);
            char* dst = (char*)&sA[r][s * NE];
#pragma unroll
            for (int q = 0; q < 4; ++q)
                async_copy16(src + q * 16, dst + q * 16);
        }
        wait_async_copies();             // this wave's weight + gather copies
        __syncthreads();
    } else {
        // Fallback: stage + convert fp32 weights in-kernel.
        const float* p0 = w0 + (size_t)g * K0 * NH;
        for (int i = tid; i < K0 * NH; i += 256) {
            int k = i / NH, n = i % NH;  // w0 is [k][n] row-major
            sW0[n][k] = (bf16)p0[i];
        }
        const float* p1 = w1 + (size_t)g * NH * NH;
        for (int i = tid; i < NH * NH; i += 256) {
            int k = i >> 7, n = i & (NH - 1);
            sW1[n][k] = (bf16)p1[i];
        }
        __syncthreads();
        for (int i = tid; i < BM * SEG; i += 256) {
            int r = i / SEG, s = i % SEG;
            const float* src = emb + (size_t)sIdx[r][s] * NE;
#pragma unroll
            for (int ec = 0; ec < NE; ++ec)
                sA[r][s * NE + ec] = (bf16)src[ec];
        }
        __syncthreads();
    }

    // Layer 0: [BM x K0] x [K0 x NH] + b0, ReLU
    gemm_layer<K0, LDK0>(&sA[0][0], K0, &sW0[0][0], sB0, &sH0[0][0], NH, wave, lane);
    __syncthreads();

    // Layer 1: [BM x NH] x [NH x NH] + b1, ReLU
    gemm_layer<NH, LDK1>(&sH0[0][0], NH, &sW1[0][0], sB1, &sH1[0][0], NH, wave, lane);
    __syncthreads();

    // Layer 2: [BM x NH] . w2 + b2, gate by smooth_z, accumulate into out.
    if (tid < BM) {
        float s = sMisc[0];
#pragma unroll 8
        for (int k = 0; k < NH; ++k)
            s += (float)sH1[tid][k] * sW2[k];
        atomicAdd(&out[rowBase + tid], sMisc[1] * s);
    }
}

// ---------------------------------------------------------------------------
// Prepass: transpose + convert one group's [K][NH] fp32 weights into the
// padded bf16 LDS image [NH][LDK].
__global__ void convert_transpose_kernel(const float* __restrict__ w,
                                         bf16* __restrict__ o,
                                         int K, int LDK) {
    const int g = blockIdx.x;
    const float* src = w + (size_t)g * K * NH;
    bf16* dst = o + (size_t)g * NH * LDK;
    for (int i = threadIdx.x; i < K * NH; i += blockDim.x) {
        int k = i / NH, n = i % NH;
        dst[n * LDK + k] = (bf16)src[i];
    }
}

__global__ void convert_emb_kernel(const float* __restrict__ e,
                                   bf16* __restrict__ o, int n) {
    int i = blockIdx.x * blockDim.x + threadIdx.x;
    if (i < n) o[i] = (bf16)e[i];
}

__global__ void zero_out_kernel(float* out, int n) {
    int i = blockIdx.x * blockDim.x + threadIdx.x;
    if (i < n) out[i] = 0.0f;
}

// ---------------------------------------------------------------------------
extern "C" void kernel_launch(void* const* d_in, const int* in_sizes, int n_in,
                              void* d_out, int out_size, void* d_ws, size_t ws_size,
                              hipStream_t stream) {
    (void)in_sizes; (void)n_in;

    const int*   mains      = (const int*)  d_in[0];
    const int*   pairs      = (const int*)  d_in[1];
    const float* emb        = (const float*)d_in[2];
    const float* mw0        = (const float*)d_in[3];
    const float* mw1        = (const float*)d_in[4];
    const float* mw2        = (const float*)d_in[5];
    const float* mb0        = (const float*)d_in[6];
    const float* mb1        = (const float*)d_in[7];
    const float* mb2        = (const float*)d_in[8];
    const float* z_main     = (const float*)d_in[9];
    const float* pw0        = (const float*)d_in[10];
    const float* pw1        = (const float*)d_in[11];
    const float* pw2        = (const float*)d_in[12];
    const float* pb0        = (const float*)d_in[13];
    const float* pb1        = (const float*)d_in[14];
    const float* pb2        = (const float*)d_in[15];
    const float* z_pairs    = (const float*)d_in[16];
    const int*   pairs_list = (const int*)  d_in[17];
    float*       out        = (float*)d_out;

    // ws layout (bf16 elements)
    const size_t n_pw0t = (size_t)NPAIR * NH * (64 + 8);
    const size_t n_pw1t = (size_t)NPAIR * NH * (NH + 8);
    const size_t n_mw0t = (size_t)NF * NH * (32 + 8);
    const size_t n_mw1t = (size_t)NF * NH * (NH + 8);
    const size_t n_embb = (size_t)EMB_ROWS * NE;
    const size_t total  = n_pw0t + n_pw1t + n_mw0t + n_mw1t + n_embb;
    const bool use_ws   = (d_ws != nullptr) && (ws_size >= total * sizeof(bf16));

    zero_out_kernel<<<dim3((out_size + 255) / 256), dim3(256), 0, stream>>>(out, out_size);

    if (use_ws) {
        bf16* pw0t = (bf16*)d_ws;
        bf16* pw1t = pw0t + n_pw0t;
        bf16* mw0t = pw1t + n_pw1t;
        bf16* mw1t = mw0t + n_mw0t;
        bf16* embb = mw1t + n_mw1t;

        convert_transpose_kernel<<<dim3(NPAIR), dim3(256), 0, stream>>>(pw0, pw0t, 64, 64 + 8);
        convert_transpose_kernel<<<dim3(NPAIR), dim3(256), 0, stream>>>(pw1, pw1t, NH, NH + 8);
        convert_transpose_kernel<<<dim3(NF),    dim3(256), 0, stream>>>(mw0, mw0t, 32, 32 + 8);
        convert_transpose_kernel<<<dim3(NF),    dim3(256), 0, stream>>>(mw1, mw1t, NH, NH + 8);
        convert_emb_kernel<<<dim3((int)((n_embb + 255) / 256)), dim3(256), 0, stream>>>(
            emb, embb, (int)n_embb);

        nam_group_kernel<NE, false, true><<<dim3(BATCH / BM, NF), dim3(256), 0, stream>>>(
            mains, pairs, pairs_list, emb, embb, mw0, mw1, mw0t, mw1t,
            mw2, mb0, mb1, mb2, z_main, out);
        nam_group_kernel<2 * NE, true, true><<<dim3(BATCH / BM, NPAIR), dim3(256), 0, stream>>>(
            mains, pairs, pairs_list, emb, embb, pw0, pw1, pw0t, pw1t,
            pw2, pb0, pb1, pb2, z_pairs, out);
    } else {
        nam_group_kernel<NE, false, false><<<dim3(BATCH / BM, NF), dim3(256), 0, stream>>>(
            mains, pairs, pairs_list, emb, nullptr, mw0, mw1, nullptr, nullptr,
            mw2, mb0, mb1, mb2, z_main, out);
        nam_group_kernel<2 * NE, true, false><<<dim3(BATCH / BM, NPAIR), dim3(256), 0, stream>>>(
            mains, pairs, pairs_list, emb, nullptr, pw0, pw1, nullptr, nullptr,
            pw2, pb0, pb1, pb2, z_pairs, out);
    }
}
